// PrefixCausalAttention_43868795961634
// MI455X (gfx1250) — compile-verified, matrix operand order checked by reading, not run on hardware
//
#include <hip/hip_runtime.h>

#define N_EMBD   1024
#define N_HEAD   16
#define HEAD_DIM 64
#define BATCH    2
#define T_GO     1024
#define T_PROT   2048
#define QKV_N    (3 * N_EMBD)

typedef __attribute__((ext_vector_type(16))) _Float16 v16h;
typedef __attribute__((ext_vector_type(2)))  _Float16 v2h;
typedef __attribute__((ext_vector_type(8)))  float    v8f;

// ---------------------------------------------------------------------------
// WMMA helper (gfx1250, wave32).  D = A(16x32 f16) * B(32x16 f16) + C(f32)
// ---------------------------------------------------------------------------
__device__ __forceinline__ v8f wmma_f16(v16h a, v16h b, v8f c) {
  return __builtin_amdgcn_wmma_f32_16x16x32_f16(
      /*neg_a=*/false, a, /*neg_b=*/false, b,
      /*c_mod=*/(short)0, c, /*reuse_a=*/false, /*reuse_b=*/false);
}

// Fragment gather from a row-major [16 x >=32] f16 tile (ld in halfs).
// ISA 7.12.2 (A / mirrored B): lanes 0-15 hold row=lane, K = 0..7 & 16..23;
// lanes 16-31 hold row=lane-16, K = 8..15 & 24..31.
// Per-lane reads are two contiguous 16B chunks -> 2x ds_load_b128.
__device__ __forceinline__ v16h ld_frag(const _Float16* p, int ld) {
  const int lane = threadIdx.x & 31;
  const _Float16* r = p + (lane & 15) * ld + ((lane >> 4) << 3);
  v16h f;
#pragma unroll
  for (int j = 0; j < 8; ++j) {
    f[j]     = r[j];
    f[8 + j] = r[16 + j];
  }
  return f;
}

__device__ __forceinline__ void st_pair(_Float16* dst, float a, float b) {
  v2h p;
  p[0] = (_Float16)a;
  p[1] = (_Float16)b;
  *reinterpret_cast<v2h*>(dst) = p;   // aligned dword store
}

// ---------------------------------------------------------------------------
// Tiled GEMM:  C[M,N] = act( A[M,K] @ W[K,N] + bias )
//   mode 0:  C = A@W + bias
//   mode 1:  C = other * sigmoid(A@W + bias) [* rowmask[row]]  (gating, in-place safe)
// BM=128 BN=128 BK=32, 256 threads (8 waves), wave tile 64x32 (4x2 WMMA tiles).
// Double-buffered LDS pipeline: next tile's global loads issue before WMMA,
// spill to the alternate buffer after compute; one barrier per K-step.
// W tile is stored TRANSPOSED in LDS so B fragments load as 2x b128 per lane.
// Staging maps each thread to contiguous LDS spans (packed v2h stores).
// ---------------------------------------------------------------------------
#define BM   128
#define BN   128
#define BK   32
#define LDA  (BK + 8)   // sA  row stride   [BM][BK+8]
#define LDBT (BK + 8)   // sBT row stride   [BN][BK+8]   (transposed W tile)

__global__ __launch_bounds__(256)
void gemm_kernel(const float* __restrict__ A, const float* __restrict__ W,
                 const float* __restrict__ bias, float* __restrict__ C,
                 const float* __restrict__ other,
                 const unsigned char* __restrict__ rowmask,
                 int M, int N, int K, int mode) {
  __shared__ _Float16 sA[2][BM * LDA];
  __shared__ _Float16 sBT[2][BN * LDBT];

  const int tid  = threadIdx.x;
  const int wave = tid >> 5;
  const int wm   = wave >> 2;   // 0..1 -> 64-row strip
  const int wn   = wave & 3;    // 0..3 -> 32-col strip
  const int bm0  = blockIdx.y * BM;
  const int bn0  = blockIdx.x * BN;

  // staging geometry:
  //  A: thread owns row (tid>>1), 16 contiguous k at (tid&1)*16
  const int arow = tid >> 1;
  const int aseg = (tid & 1) << 4;
  //  B: 8 (k,k+1) pairs; pair pp = p*256+tid -> n = pp&127, k = 2*(pp>>7)
  const int bn_  = tid & 127;
  const int bkh  = tid >> 7;         // 0..1

  v8f acc[4][2] = {};

  // ---- prologue: stage tile 0 ----
  {
    const float* ap = A + (size_t)(bm0 + arow) * K + aseg;
    _Float16*    dp = &sA[0][arow * LDA + aseg];
#pragma unroll
    for (int j = 0; j < 8; ++j) st_pair(dp + 2 * j, ap[2 * j], ap[2 * j + 1]);
#pragma unroll
    for (int p = 0; p < 8; ++p) {
      int k = (bkh + 2 * p) * 2;     // 0,2,...,30
      st_pair(&sBT[0][bn_ * LDBT + k],
              W[(size_t)k * N + bn0 + bn_],
              W[(size_t)(k + 1) * N + bn0 + bn_]);
    }
  }
  __syncthreads();

  const int nsteps = K / BK;
  for (int s = 0; s < nsteps; ++s) {
    const int cur   = s & 1;
    const int nxt   = cur ^ 1;
    const bool more = (s + 1) < nsteps;

    // issue next tile's global loads first (latency hides behind WMMA)
    float ra[16], rb0[8], rb1[8];
    if (more) {
      const int k0 = (s + 1) * BK;
      const float* ap = A + (size_t)(bm0 + arow) * K + k0 + aseg;
#pragma unroll
      for (int j = 0; j < 16; ++j) ra[j] = ap[j];
#pragma unroll
      for (int p = 0; p < 8; ++p) {
        int k = (bkh + 2 * p) * 2;
        rb0[p] = W[(size_t)(k0 + k) * N + bn0 + bn_];
        rb1[p] = W[(size_t)(k0 + k + 1) * N + bn0 + bn_];
      }
    }

    // compute on current buffer
    v16h fb[2];
#pragma unroll
    for (int j = 0; j < 2; ++j)
      fb[j] = ld_frag(&sBT[cur][(wn * 32 + j * 16) * LDBT], LDBT);
#pragma unroll
    for (int i = 0; i < 4; ++i) {
      v16h fa = ld_frag(&sA[cur][(wm * 64 + i * 16) * LDA], LDA);
#pragma unroll
      for (int j = 0; j < 2; ++j)
        acc[i][j] = wmma_f16(fa, fb[j], acc[i][j]);
    }

    // drain staged registers into the alternate buffer (packed dword stores)
    if (more) {
      _Float16* dp = &sA[nxt][arow * LDA + aseg];
#pragma unroll
      for (int j = 0; j < 8; ++j) st_pair(dp + 2 * j, ra[2 * j], ra[2 * j + 1]);
#pragma unroll
      for (int p = 0; p < 8; ++p) {
        int k = (bkh + 2 * p) * 2;
        st_pair(&sBT[nxt][bn_ * LDBT + k], rb0[p], rb1[p]);
      }
    }
    __syncthreads();
  }

  // Epilogue.  C/D layout: element (m, n): m = (lane>>4)*8 + r, n = lane&15.
  const int lane = tid & 31;
  const int ncol = lane & 15;
  const int mrow = (lane >> 4) << 3;
#pragma unroll
  for (int i = 0; i < 4; ++i) {
#pragma unroll
    for (int j = 0; j < 2; ++j) {
#pragma unroll
      for (int r = 0; r < 8; ++r) {
        int row = bm0 + wm * 64 + i * 16 + mrow + r;
        int col = bn0 + wn * 32 + j * 16 + ncol;
        float v = acc[i][j][r] + bias[col];
        if (mode == 1) {
          float g = 1.f / (1.f + __expf(-v));
          v = other[(size_t)row * N + col] * g;
          if (rowmask) v *= (float)rowmask[row];
        }
        C[(size_t)row * N + col] = v;
      }
    }
  }
}

// ---------------------------------------------------------------------------
// Flash attention over the prefix-causal key space.
//  qmode 0: protein queries -> keys = protein only, mask = protein_mask
//  qmode 1: go queries      -> keys = [protein | go], protein_mask on prefix,
//                              (go_mask & causal) on go keys
// Block = 128 threads (4 waves); block owns 64 queries of one (b,h);
// each wave owns a 16-query strip. 32-key tiles, f16 WMMA, f32 softmax.
// K tile is key-major (QK^T B-fragments read contiguous d);
// V tile is staged TRANSPOSED (d-major) so PV B-fragments read contiguous k.
// ---------------------------------------------------------------------------
#define KT   32
#define LDK  (HEAD_DIM + 8)  // sQ/sK row stride
#define LDVT (KT + 8)        // sVT row stride (transposed V)
#define LDP  (KT + 8)        // sP row stride

__global__ __launch_bounds__(128)
void attn_kernel(const float* __restrict__ qkv_q,   // [B*Tq][3072]
                 const float* __restrict__ qkv_pr,  // [B*Tp][3072]
                 const float* __restrict__ qkv_go,  // [B*Tg][3072]
                 const unsigned char* __restrict__ pmask,  // [B*Tp]
                 const unsigned char* __restrict__ gmask,  // [B*Tg]
                 float* __restrict__ out,           // [B*Tq][1024]
                 int Tq, int qmode) {
  __shared__ _Float16 sQ[64 * LDK];
  __shared__ _Float16 sK[KT * LDK];
  __shared__ _Float16 sVT[HEAD_DIM * LDVT];
  __shared__ _Float16 sP[4][16 * LDP];

  const int tid  = threadIdx.x;
  const int lane = tid & 31;
  const int wv   = tid >> 5;          // wave -> query rows [qb+16*wv, +16)
  const int qb   = blockIdx.x * 64;
  const int h    = blockIdx.y;
  const int b    = blockIdx.z;

  // Stage Q tile (64x64) as f16; thread owns row (tid>>1), 32 contiguous d.
  {
    const int r = tid >> 1, dseg = (tid & 1) << 5;
    const int q = qb + r;
    float qm = (qmode == 1) ? (float)gmask[b * T_GO + q] : 1.f;
    const float* qp = qkv_q + (size_t)(b * Tq + q) * QKV_N + h * HEAD_DIM + dseg;
    _Float16*    dp = &sQ[r * LDK + dseg];
#pragma unroll
    for (int j = 0; j < 16; ++j)
      st_pair(dp + 2 * j, qp[2 * j] * qm, qp[2 * j + 1] * qm);
  }
  __syncthreads();

  const v16h fq0 = ld_frag(&sQ[(wv * 16) * LDK + 0],  LDK);
  const v16h fq1 = ld_frag(&sQ[(wv * 16) * LDK + 32], LDK);

  v8f o[4] = {};
  float mrow[8], lrow[8];
#pragma unroll
  for (int r = 0; r < 8; ++r) { mrow[r] = -1e30f; lrow[r] = 0.f; }

  const int nk = (qmode == 0) ? T_PROT : (T_PROT + qb + 64);  // causal frontier

  // staging geometry (tiles never straddle the prot/go seam):
  //  K: thread owns key row (tid>>2), 16 contiguous d at (tid&3)*16
  const int kkr  = tid >> 2;
  const int kds  = (tid & 3) << 4;
  //  V: 8 (kr,kr+1) pairs; pair pp = p*128+tid -> d = pp&63, kr = 2*(pp>>6)
  const int vd   = tid & 63;
  const int vkh  = tid >> 6;          // 0..1

  for (int kb0 = 0; kb0 < nk; kb0 += KT) {
    __syncthreads();  // retire readers of previous sK/sVT
    {
      // K rows: contiguous halves per thread
      int j = kb0 + kkr;
      const float* srck =
          (j < T_PROT)
              ? qkv_pr + (size_t)(b * T_PROT + j) * QKV_N + N_EMBD + h * HEAD_DIM + kds
              : qkv_go + (size_t)(b * T_GO + (j - T_PROT)) * QKV_N + N_EMBD + h * HEAD_DIM + kds;
      _Float16* dp = &sK[kkr * LDK + kds];
#pragma unroll
      for (int e = 0; e < 8; ++e) st_pair(dp + 2 * e, srck[2 * e], srck[2 * e + 1]);
      // V transposed: paired-kr dword stores
#pragma unroll
      for (int p = 0; p < 8; ++p) {
        int kr = (vkh + 2 * p) * 2;   // 0,2,...,30
        int j0 = kb0 + kr;
        const float* s0 =
            (j0 < T_PROT)
                ? qkv_pr + (size_t)(b * T_PROT + j0) * QKV_N + 2 * N_EMBD + h * HEAD_DIM + vd
                : qkv_go + (size_t)(b * T_GO + (j0 - T_PROT)) * QKV_N + 2 * N_EMBD + h * HEAD_DIM + vd;
        st_pair(&sVT[vd * LDVT + kr], s0[0], s0[QKV_N]);  // rows j0, j0+1
      }
    }
    __syncthreads();

    // S = (Q . K^T) * 1/sqrt(64), two 16x16 tiles per wave, masked.
    float s[2][8];
#pragma unroll
    for (int nt = 0; nt < 2; ++nt) {
      v8f sc = {};
      sc = wmma_f16(fq0, ld_frag(&sK[(nt * 16) * LDK + 0],  LDK), sc);
      sc = wmma_f16(fq1, ld_frag(&sK[(nt * 16) * LDK + 32], LDK), sc);

      const int jcol = kb0 + nt * 16 + (lane & 15);
      const int jg   = jcol - T_PROT;
      float kvalid = (qmode == 0 || jcol < T_PROT)
                         ? (float)pmask[b * T_PROT + jcol]
                         : (float)gmask[b * T_GO + jg];
#pragma unroll
      for (int r = 0; r < 8; ++r) {
        int qrow = qb + wv * 16 + ((lane >> 4) << 3) + r;
        bool ok = kvalid > 0.f;
        if (qmode == 1 && jcol >= T_PROT) ok = ok && (jg <= qrow);
        s[nt][r] = ok ? sc[r] * 0.125f : -3.0e38f;
      }
    }

    // Online softmax: row = 16 lanes of a half-wave -> width-16 reductions.
    float mnew[8], alpha[8], rs[8];
#pragma unroll
    for (int r = 0; r < 8; ++r) {
      float t = fmaxf(s[0][r], s[1][r]);
#pragma unroll
      for (int off = 8; off >= 1; off >>= 1)
        t = fmaxf(t, __shfl_xor(t, off, 16));
      mnew[r]  = fmaxf(mrow[r], t);
      alpha[r] = __expf(mrow[r] - mnew[r]);
      rs[r]    = 0.f;
    }
#pragma unroll
    for (int nt = 0; nt < 2; ++nt) {
#pragma unroll
      for (int r = 0; r < 8; ++r) {
        float p = __expf(s[nt][r] - mnew[r]);
        rs[r] += p;
        sP[wv][(((lane >> 4) << 3) + r) * LDP + nt * 16 + (lane & 15)] = (_Float16)p;
      }
    }
#pragma unroll
    for (int r = 0; r < 8; ++r) {
#pragma unroll
      for (int off = 8; off >= 1; off >>= 1)
        rs[r] += __shfl_xor(rs[r], off, 16);
      lrow[r] = lrow[r] * alpha[r] + rs[r];
      mrow[r] = mnew[r];
    }
#pragma unroll
    for (int dt = 0; dt < 4; ++dt)
#pragma unroll
      for (int r = 0; r < 8; ++r) o[dt][r] *= alpha[r];

    // sP is per-wave; same-wave DS ops are in-order (ISA 7.1) and the loads
    // below alias the stores above, so no workgroup barrier is needed here.

    // O += P(16x32) @ V(32x64)
    const v16h fp = ld_frag(&sP[wv][0], LDP);
#pragma unroll
    for (int dt = 0; dt < 4; ++dt)
      o[dt] = wmma_f16(fp, ld_frag(&sVT[(dt * 16) * LDVT], LDVT), o[dt]);
  }

  // Normalize + merge heads; zero fully-masked rows; go output masked by go_mask.
  const int ncol = lane & 15;
  const int mr   = (lane >> 4) << 3;
#pragma unroll
  for (int r = 0; r < 8; ++r) {
    int q = qb + wv * 16 + mr + r;
    float inv = (lrow[r] > 0.f) ? 1.f / lrow[r] : 0.f;
    float qm  = (qmode == 1) ? (float)gmask[b * T_GO + q] : 1.f;
#pragma unroll
    for (int dt = 0; dt < 4; ++dt)
      out[(size_t)(b * Tq + q) * N_EMBD + h * HEAD_DIM + dt * 16 + ncol] =
          o[dt][r] * inv * qm;
  }
}

// ---------------------------------------------------------------------------
extern "C" void kernel_launch(void* const* d_in, const int* in_sizes, int n_in,
                              void* d_out, int out_size, void* d_ws, size_t ws_size,
                              hipStream_t stream) {
  const float* go_states = (const float*)d_in[0];
  const float* pr_states = (const float*)d_in[1];
  const unsigned char* pmask = (const unsigned char*)d_in[2];
  const unsigned char* gmask = (const unsigned char*)d_in[3];
  const float* go_qkv_w  = (const float*)d_in[4];
  const float* go_qkv_b  = (const float*)d_in[5];
  const float* pr_qkv_w  = (const float*)d_in[6];
  const float* pr_qkv_b  = (const float*)d_in[7];
  const float* go_proj_w = (const float*)d_in[8];
  const float* go_proj_b = (const float*)d_in[9];
  const float* pr_proj_w = (const float*)d_in[10];
  const float* pr_proj_b = (const float*)d_in[11];
  const float* go_gate_w = (const float*)d_in[12];
  const float* go_gate_b = (const float*)d_in[13];
  const float* pr_gate_w = (const float*)d_in[14];
  const float* pr_gate_b = (const float*)d_in[15];

  float* ws      = (float*)d_ws;
  float* qkv_go  = ws;                                      // B*Tg*3072
  float* qkv_pr  = qkv_go + (size_t)BATCH * T_GO * QKV_N;   // B*Tp*3072
  float* attn_pr = qkv_pr + (size_t)BATCH * T_PROT * QKV_N; // B*Tp*1024
  float* attn_go = attn_pr + (size_t)BATCH * T_PROT * N_EMBD;

  float* out_pr = (float*)d_out;
  float* out_go = out_pr + (size_t)BATCH * T_PROT * N_EMBD;

  const dim3 gblk(256);

  // 1) QKV projections
  gemm_kernel<<<dim3(QKV_N / BN, (BATCH * T_GO) / BM), gblk, 0, stream>>>(
      go_states, go_qkv_w, go_qkv_b, qkv_go, nullptr, nullptr,
      BATCH * T_GO, QKV_N, N_EMBD, 0);
  gemm_kernel<<<dim3(QKV_N / BN, (BATCH * T_PROT) / BM), gblk, 0, stream>>>(
      pr_states, pr_qkv_w, pr_qkv_b, qkv_pr, nullptr, nullptr,
      BATCH * T_PROT, QKV_N, N_EMBD, 0);

  // 2) attention (protein queries: protein keys only; go queries: prefix-causal)
  attn_kernel<<<dim3(T_PROT / 64, N_HEAD, BATCH), dim3(128), 0, stream>>>(
      qkv_pr, qkv_pr, qkv_go, pmask, gmask, attn_pr, T_PROT, 0);
  attn_kernel<<<dim3(T_GO / 64, N_HEAD, BATCH), dim3(128), 0, stream>>>(
      qkv_go, qkv_pr, qkv_go, pmask, gmask, attn_go, T_GO, 1);

  // 3) gating (in-place: attn *= sigmoid(states@gate_w + b) [* go_mask])
  gemm_kernel<<<dim3(N_EMBD / BN, (BATCH * T_PROT) / BM), gblk, 0, stream>>>(
      pr_states, pr_gate_w, pr_gate_b, attn_pr, attn_pr, nullptr,
      BATCH * T_PROT, N_EMBD, N_EMBD, 1);
  gemm_kernel<<<dim3(N_EMBD / BN, (BATCH * T_GO) / BM), gblk, 0, stream>>>(
      go_states, go_gate_w, go_gate_b, attn_go, attn_go, gmask,
      BATCH * T_GO, N_EMBD, N_EMBD, 1);

  // 4) output projections (tuple order: protein, go)
  gemm_kernel<<<dim3(N_EMBD / BN, (BATCH * T_PROT) / BM), gblk, 0, stream>>>(
      attn_pr, pr_proj_w, pr_proj_b, out_pr, nullptr, nullptr,
      BATCH * T_PROT, N_EMBD, N_EMBD, 0);
  gemm_kernel<<<dim3(N_EMBD / BN, (BATCH * T_GO) / BM), gblk, 0, stream>>>(
      attn_go, go_proj_w, go_proj_b, out_go, nullptr, nullptr,
      BATCH * T_GO, N_EMBD, N_EMBD, 0);
}
